// MoELayer_61564061221293
// MI455X (gfx1250) — matrix operand order repaired and checked by reading, not compiled, not run on hardware
//
#include <hip/hip_runtime.h>
#include <math.h>

// ---------------- problem constants ----------------
#define T_TOK 8192
#define HID   1024
#define NEXP  8
#define INTER 2816

// ---------------- WMMA types ----------------
typedef __bf16 v16bf __attribute__((ext_vector_type(16)));
typedef float  v8f   __attribute__((ext_vector_type(8)));

union FragAB { uint4 u[2]; v16bf v; };
union Pack8  { uint4 u; __bf16 h[8]; };

// ---------------- workspace layout (bytes) ----------------
static constexpr size_t XB_OFF  = 0;                                   // x bf16   [T][H]
static constexpr size_t XB_SZ   = (size_t)T_TOK * HID * 2;
static constexpr size_t WG_OFF  = XB_OFF + XB_SZ;                      // w_gate^T bf16 [E][I][H]
static constexpr size_t WGT_SZ  = (size_t)NEXP * HID * INTER * 2;
static constexpr size_t WU_OFF  = WG_OFF + WGT_SZ;                     // w_up^T   bf16 [E][I][H]
static constexpr size_t WD_OFF  = WU_OFF + WGT_SZ;                     // w_down^T bf16 [E][H][I]
static constexpr size_t HB_OFF  = WD_OFF + WGT_SZ;                     // h bf16 [T][I] (per-expert reuse)
static constexpr size_t HB_SZ   = (size_t)T_TOK * INTER * 2;
static constexpr size_t CNT_OFF = HB_OFF + HB_SZ;                      // int counts[E]
static constexpr size_t IDX_OFF = CNT_OFF + 256;                       // int idx[E][T]
static constexpr size_t WTS_OFF = IDX_OFF + (size_t)NEXP * T_TOK * 4;  // float wts[E][T]

// -------- CDNA5 async global->LDS copy (16B per lane, ASYNCcnt-tracked) --------
__device__ __forceinline__ void async_copy16(unsigned lds_off, const void* g) {
  asm volatile("global_load_async_to_lds_b128 %0, %1, off"
               :: "v"(lds_off), "v"(g) : "memory");
}

// ---------------- init: zero output + counters ----------------
__global__ __launch_bounds__(256) void init_kernel(float* __restrict__ out,
                                                   int* __restrict__ counts) {
  long i = ((long)blockIdx.x * 256 + threadIdx.x) * 4;
  if (i < (long)T_TOK * HID) {
    float4 z; z.x = 0.f; z.y = 0.f; z.z = 0.f; z.w = 0.f;
    *(float4*)(out + i) = z;
  }
  if (blockIdx.x == 0 && threadIdx.x < NEXP) counts[threadIdx.x] = 0;
}

// ---------------- f32 -> bf16 straight conversion (x) ----------------
__global__ __launch_bounds__(256) void cvt_f32_bf16(const float* __restrict__ s,
                                                    __bf16* __restrict__ d, long n) {
  long i = ((long)blockIdx.x * 256 + threadIdx.x) * 8;
  if (i + 8 > n) return;
  const float4* sp = (const float4*)(s + i);
  float4 a = sp[0], b = sp[1];
  Pack8 p;
  p.h[0] = (__bf16)a.x; p.h[1] = (__bf16)a.y; p.h[2] = (__bf16)a.z; p.h[3] = (__bf16)a.w;
  p.h[4] = (__bf16)b.x; p.h[5] = (__bf16)b.y; p.h[6] = (__bf16)b.z; p.h[7] = (__bf16)b.w;
  *(uint4*)(d + i) = p.u;
}

// ------- f32 [E][K][N] -> bf16 [E][N][K] tiled transpose-convert (64x64 tiles) -------
__global__ __launch_bounds__(256) void cvt_transpose(const float* __restrict__ s,
                                                     __bf16* __restrict__ d,
                                                     int K, int N) {
  __shared__ __bf16 tile[64][72];                 // [n][k], padded
  const size_t eoff = (size_t)blockIdx.z * K * N;
  const int kb = blockIdx.x * 64, nb = blockIdx.y * 64;
  const int tid = threadIdx.x;
#pragma unroll
  for (int i = 0; i < 4; ++i) {                   // read 64x64 f32 coalesced
    int id = tid + 256 * i;
    int kr = id >> 4;                              // 0..63
    int nc = (id & 15) * 4;                        // 0..60
    float4 v = *(const float4*)(s + eoff + (size_t)(kb + kr) * N + (nb + nc));
    tile[nc + 0][kr] = (__bf16)v.x;
    tile[nc + 1][kr] = (__bf16)v.y;
    tile[nc + 2][kr] = (__bf16)v.z;
    tile[nc + 3][kr] = (__bf16)v.w;
  }
  __syncthreads();
#pragma unroll
  for (int i = 0; i < 2; ++i) {                   // write 64x64 bf16 coalesced
    int id = tid + 256 * i;
    int nr = id >> 3;                              // 0..63
    int kc = (id & 7) * 8;                         // 0..56
    Pack8 p;
#pragma unroll
    for (int j = 0; j < 8; ++j) p.h[j] = tile[nr][kc + j];
    *(uint4*)(d + eoff + (size_t)(nb + nr) * K + (kb + kc)) = p.u;
  }
}

// ---------------- router: one wave(32) per token ----------------
__global__ __launch_bounds__(256) void router_kernel(const float* __restrict__ x,
                                                     const float* __restrict__ wr,
                                                     int* __restrict__ counts,
                                                     int* __restrict__ idx,
                                                     float* __restrict__ wts) {
  int t    = blockIdx.x * 8 + (threadIdx.x >> 5);
  int lane = threadIdx.x & 31;
  float acc[NEXP];
#pragma unroll
  for (int e = 0; e < NEXP; ++e) acc[e] = 0.f;
  const float* xr = x + (size_t)t * HID;
  for (int h = lane; h < HID; h += 32) {
    float xv = xr[h];
#pragma unroll
    for (int e = 0; e < NEXP; ++e) acc[e] += xv * wr[e * HID + h];
  }
#pragma unroll
  for (int e = 0; e < NEXP; ++e) {
#pragma unroll
    for (int off = 16; off > 0; off >>= 1) acc[e] += __shfl_xor(acc[e], off);
  }
  if (lane == 0) {
    int e0 = 0; float l0 = acc[0];
#pragma unroll
    for (int e = 1; e < NEXP; ++e) if (acc[e] > l0) { l0 = acc[e]; e0 = e; }
    int e1 = -1; float l1 = -3.4e38f;
#pragma unroll
    for (int e = 0; e < NEXP; ++e) if (e != e0 && acc[e] > l1) { l1 = acc[e]; e1 = e; }
    float m  = fmaxf(l0, l1);
    float g0 = __expf(l0 - m), g1 = __expf(l1 - m);
    float s  = g0 + g1; g0 /= s; g1 /= s;
    int p0 = atomicAdd(&counts[e0], 1);
    idx[e0 * T_TOK + p0] = t; wts[e0 * T_TOK + p0] = g0;
    int p1 = atomicAdd(&counts[e1], 1);
    idx[e1 * T_TOK + p1] = t; wts[e1 * T_TOK + p1] = g1;
  }
}

// ---------------- fused gate+up GEMM + SiLU epilogue ----------------
// Tile 128 tokens x 128 inter; K-loop over H step 32; async double-buffered LDS.
// 8 waves as 2(M) x 4(N); each wave 64x32 -> 4x2 C frags per matrix.
__global__ __launch_bounds__(256) void gateup_kernel(int e,
    const __bf16* __restrict__ xb, const __bf16* __restrict__ wgT,
    const __bf16* __restrict__ wuT, __bf16* __restrict__ hb,
    const int* __restrict__ counts, const int* __restrict__ idx) {
  const int cnt = counts[e];
  const int m0  = blockIdx.x * 128;
  if (m0 >= cnt) return;
  const int n0  = blockIdx.y * 128;

  // [buf][A|G|U][128 rows][32 k] bf16 : 2 x 24KB
  __shared__ __align__(16) unsigned char lds[2 * 24 * 1024];

  const int tid  = threadIdx.x;
  const int lane = tid & 31;
  const int wave = tid >> 5;
  const int wm   = wave & 1;
  const int wn   = wave >> 1;
  const int lh   = lane & 15;
  const int hi   = lane >> 4;
  const int koffA = hi * 32;   // A frag: byte offset of K half per lane half
  const int koffB = hi * 32;   // B frag: byte offset of K half per lane half

  // tile loader: 512 16B chunks per 8KB tile, 2 per thread (rows row0,row1)
  const int row0 = tid >> 2, part = tid & 3;
  const int row1 = (tid + 256) >> 2;
  const int s0 = m0 + row0, s1 = m0 + row1;
  const int tok0 = (s0 < cnt) ? idx[e * T_TOK + s0] : 0;
  const int tok1 = (s1 < cnt) ? idx[e * T_TOK + s1] : 0;
  const __bf16* xrow0 = xb + (size_t)tok0 * HID;
  const __bf16* xrow1 = xb + (size_t)tok1 * HID;
  const __bf16* grow0 = wgT + ((size_t)e * INTER + (n0 + row0)) * HID;
  const __bf16* grow1 = wgT + ((size_t)e * INTER + (n0 + row1)) * HID;
  const __bf16* urow0 = wuT + ((size_t)e * INTER + (n0 + row0)) * HID;
  const __bf16* urow1 = wuT + ((size_t)e * INTER + (n0 + row1)) * HID;

  const unsigned ldsBase = (unsigned)(size_t)(void*)lds;
  const unsigned c0 = (unsigned)((row0 * 32 + part * 8) * 2);
  const unsigned c1 = (unsigned)((row1 * 32 + part * 8) * 2);

  auto issue = [&](int k0, int buf) {
    unsigned b = ldsBase + (unsigned)buf * 24576u;
    async_copy16(b + c0,          xrow0 + k0 + part * 8);
    async_copy16(b + c1,          xrow1 + k0 + part * 8);
    async_copy16(b + 8192u + c0,  grow0 + k0 + part * 8);
    async_copy16(b + 8192u + c1,  grow1 + k0 + part * 8);
    async_copy16(b + 16384u + c0, urow0 + k0 + part * 8);
    async_copy16(b + 16384u + c1, urow1 + k0 + part * 8);
  };

  v8f cg[4][2], cu[4][2];
#pragma unroll
  for (int mi = 0; mi < 4; ++mi)
#pragma unroll
    for (int ni = 0; ni < 2; ++ni) { cg[mi][ni] = (v8f)0.f; cu[mi][ni] = (v8f)0.f; }

  const int NK = HID / 32;
  issue(0, 0);
  for (int kk = 0; kk < NK; ++kk) {
    const int cur = kk & 1;
    if (kk + 1 < NK) {
      issue((kk + 1) * 32, cur ^ 1);
      asm volatile("s_wait_asynccnt 0x6" ::: "memory"); // current stage resident
    } else {
      asm volatile("s_wait_asynccnt 0x0" ::: "memory");
    }
    __syncthreads();

    const unsigned char* l = lds + cur * 24576;
    FragAB a[4], bg[2], bu[2];
#pragma unroll
    for (int mi = 0; mi < 4; ++mi) {
      const unsigned char* p = l + (wm * 64 + mi * 16 + lh) * 64 + koffA;
      a[mi].u[0] = *(const uint4*)(p);
      a[mi].u[1] = *(const uint4*)(p + 16);
    }
#pragma unroll
    for (int ni = 0; ni < 2; ++ni) {
      const unsigned char* pg = l + 8192 + (wn * 32 + ni * 16 + lh) * 64 + koffB;
      bg[ni].u[0] = *(const uint4*)(pg);
      bg[ni].u[1] = *(const uint4*)(pg + 16);
      const unsigned char* pu = l + 16384 + (wn * 32 + ni * 16 + lh) * 64 + koffB;
      bu[ni].u[0] = *(const uint4*)(pu);
      bu[ni].u[1] = *(const uint4*)(pu + 16);
    }
#pragma unroll
    for (int mi = 0; mi < 4; ++mi) {
#pragma unroll
      for (int ni = 0; ni < 2; ++ni) {
        cg[mi][ni] = __builtin_amdgcn_wmma_f32_16x16x32_bf16(
            false, a[mi].v, false, bg[ni].v, (short)0, cg[mi][ni], false, false);
        cu[mi][ni] = __builtin_amdgcn_wmma_f32_16x16x32_bf16(
            false, a[mi].v, false, bu[ni].v, (short)0, cu[mi][ni], false, false);
      }
    }
    __syncthreads();  // all waves done reading buf before it is re-filled
  }

  // epilogue: h = silu(g)*u  (zero rows past count so down-GEMM reads clean data)
#pragma unroll
  for (int mi = 0; mi < 4; ++mi) {
#pragma unroll
    for (int ni = 0; ni < 2; ++ni) {
      int col = n0 + wn * 32 + ni * 16 + lh;
#pragma unroll
      for (int j = 0; j < 8; ++j) {
        int slot = m0 + wm * 64 + mi * 16 + j + 8 * hi;
        float g = cg[mi][ni][j], u = cu[mi][ni][j];
        float hval = (slot < cnt) ? (g / (1.f + __expf(-g))) * u : 0.f;
        hb[(size_t)slot * INTER + col] = (__bf16)hval;
      }
    }
  }
}

// ---------------- down GEMM + weighted scatter-accumulate ----------------
__global__ __launch_bounds__(256) void down_kernel(int e,
    const __bf16* __restrict__ hb, const __bf16* __restrict__ wdT,
    float* __restrict__ out, const int* __restrict__ counts,
    const int* __restrict__ idx, const float* __restrict__ wts) {
  const int cnt = counts[e];
  const int m0  = blockIdx.x * 128;
  if (m0 >= cnt) return;
  const int n0  = blockIdx.y * 128;

  // [buf][A|B][128 rows][32 k] bf16 : 2 x 16KB
  __shared__ __align__(16) unsigned char lds[2 * 16 * 1024];

  const int tid  = threadIdx.x;
  const int lane = tid & 31;
  const int wave = tid >> 5;
  const int wm   = wave & 1;
  const int wn   = wave >> 1;
  const int lh   = lane & 15;
  const int hi   = lane >> 4;
  const int koffA = hi * 32;
  const int koffB = hi * 32;

  const int row0 = tid >> 2, part = tid & 3;
  const int row1 = (tid + 256) >> 2;
  const __bf16* arow0 = hb + (size_t)(m0 + row0) * INTER;
  const __bf16* arow1 = hb + (size_t)(m0 + row1) * INTER;
  const __bf16* brow0 = wdT + ((size_t)e * HID + (n0 + row0)) * INTER;
  const __bf16* brow1 = wdT + ((size_t)e * HID + (n0 + row1)) * INTER;

  const unsigned ldsBase = (unsigned)(size_t)(void*)lds;
  const unsigned c0 = (unsigned)((row0 * 32 + part * 8) * 2);
  const unsigned c1 = (unsigned)((row1 * 32 + part * 8) * 2);

  auto issue = [&](int k0, int buf) {
    unsigned b = ldsBase + (unsigned)buf * 16384u;
    async_copy16(b + c0,         arow0 + k0 + part * 8);
    async_copy16(b + c1,         arow1 + k0 + part * 8);
    async_copy16(b + 8192u + c0, brow0 + k0 + part * 8);
    async_copy16(b + 8192u + c1, brow1 + k0 + part * 8);
  };

  v8f c[4][2];
#pragma unroll
  for (int mi = 0; mi < 4; ++mi)
#pragma unroll
    for (int ni = 0; ni < 2; ++ni) c[mi][ni] = (v8f)0.f;

  const int NK = INTER / 32;
  issue(0, 0);
  for (int kk = 0; kk < NK; ++kk) {
    const int cur = kk & 1;
    if (kk + 1 < NK) {
      issue((kk + 1) * 32, cur ^ 1);
      asm volatile("s_wait_asynccnt 0x4" ::: "memory");
    } else {
      asm volatile("s_wait_asynccnt 0x0" ::: "memory");
    }
    __syncthreads();

    const unsigned char* l = lds + cur * 16384;
    FragAB a[4], b[2];
#pragma unroll
    for (int mi = 0; mi < 4; ++mi) {
      const unsigned char* p = l + (wm * 64 + mi * 16 + lh) * 64 + koffA;
      a[mi].u[0] = *(const uint4*)(p);
      a[mi].u[1] = *(const uint4*)(p + 16);
    }
#pragma unroll
    for (int ni = 0; ni < 2; ++ni) {
      const unsigned char* p = l + 8192 + (wn * 32 + ni * 16 + lh) * 64 + koffB;
      b[ni].u[0] = *(const uint4*)(p);
      b[ni].u[1] = *(const uint4*)(p + 16);
    }
#pragma unroll
    for (int mi = 0; mi < 4; ++mi)
#pragma unroll
      for (int ni = 0; ni < 2; ++ni)
        c[mi][ni] = __builtin_amdgcn_wmma_f32_16x16x32_bf16(
            false, a[mi].v, false, b[ni].v, (short)0, c[mi][ni], false, false);
    __syncthreads();
  }

  // weighted scatter: out[t, col] += w * acc (race-free: experts serialized;
  // each (slot,col) owned by exactly one thread in this grid)
#pragma unroll
  for (int mi = 0; mi < 4; ++mi) {
#pragma unroll
    for (int j = 0; j < 8; ++j) {
      int slot = m0 + wm * 64 + mi * 16 + j + 8 * hi;
      if (slot < cnt) {
        int   t = idx[e * T_TOK + slot];
        float w = wts[e * T_TOK + slot];
#pragma unroll
        for (int ni = 0; ni < 2; ++ni) {
          int col = n0 + wn * 32 + ni * 16 + lh;
          float* o = out + (size_t)t * HID + col;
          *o += w * c[mi][ni][j];
        }
      }
    }
  }
}

// ---------------- host launcher ----------------
extern "C" void kernel_launch(void* const* d_in, const int* in_sizes, int n_in,
                              void* d_out, int out_size, void* d_ws, size_t ws_size,
                              hipStream_t stream) {
  const float* x  = (const float*)d_in[0];
  const float* wr = (const float*)d_in[1];
  const float* wg = (const float*)d_in[2];
  const float* wu = (const float*)d_in[3];
  const float* wd = (const float*)d_in[4];
  float* out = (float*)d_out;

  unsigned char* ws = (unsigned char*)d_ws;
  __bf16* xb  = (__bf16*)(ws + XB_OFF);
  __bf16* wgT = (__bf16*)(ws + WG_OFF);
  __bf16* wuT = (__bf16*)(ws + WU_OFF);
  __bf16* wdT = (__bf16*)(ws + WD_OFF);
  __bf16* hb  = (__bf16*)(ws + HB_OFF);
  int*    cnt = (int*)(ws + CNT_OFF);
  int*    idx = (int*)(ws + IDX_OFF);
  float*  wts = (float*)(ws + WTS_OFF);

  // 1) zero output + counters
  init_kernel<<<(T_TOK * HID / 4 + 255) / 256, 256, 0, stream>>>(out, cnt);

  // 2) conversions: x straight; weights transpose-converted to K-contiguous layouts
  const long nx = (long)T_TOK * HID;
  cvt_f32_bf16<<<(unsigned)(nx / 8 / 256), 256, 0, stream>>>(x, xb, nx);
  cvt_transpose<<<dim3(HID / 64, INTER / 64, NEXP), 256, 0, stream>>>(wg, wgT, HID, INTER);
  cvt_transpose<<<dim3(HID / 64, INTER / 64, NEXP), 256, 0, stream>>>(wu, wuT, HID, INTER);
  cvt_transpose<<<dim3(INTER / 64, HID / 64, NEXP), 256, 0, stream>>>(wd, wdT, INTER, HID);

  // 3) router + compaction
  router_kernel<<<T_TOK / 8, 256, 0, stream>>>(x, wr, cnt, idx, wts);

  // 4) per-expert GEMMs (sequential -> h reuse + race-free out accumulation)
  for (int e = 0; e < NEXP; ++e) {
    gateup_kernel<<<dim3(T_TOK / 128, INTER / 128), 256, 0, stream>>>(
        e, xb, wgT, wuT, hb, cnt, idx);
    down_kernel<<<dim3(T_TOK / 128, HID / 128), 256, 0, stream>>>(
        e, hb, wdT, out, cnt, idx, wts);
  }
}
